// ResearchGNN_35983236006564
// MI455X (gfx1250) — matrix operand order repaired and checked by reading, not compiled, not run on hardware
//
#include <hip/hip_runtime.h>
#include <hip/hip_bf16.h>

#define NN 100000
#define EE 1600000
#define HH 128
#define LL 4
#define GG 128
#define CC 5
#define EPS_BN 1e-5f

typedef __attribute__((ext_vector_type(16))) _Float16 v16h;
typedef __attribute__((ext_vector_type(8)))  float    v8f;

__device__ __forceinline__ void atomAddF(float* p, float v) {
    (void)__hip_atomic_fetch_add(p, v, __ATOMIC_RELAXED, __HIP_MEMORY_SCOPE_AGENT);
}

// ---------------- utility ----------------
__global__ void k_zero(float* __restrict__ p, int n) {
    int i = blockIdx.x * blockDim.x + threadIdx.x;
    if (i < n) p[i] = 0.0f;
}

// deg starts at 1.0 (self loop), then count incoming edges
__global__ void k_init_deg(float* __restrict__ deg) {
    int i = blockIdx.x * blockDim.x + threadIdx.x;
    if (i < NN) deg[i] = 1.0f;
}

__global__ void k_count_deg(const int* __restrict__ dst, float* __restrict__ deg) {
    int e = blockIdx.x * blockDim.x + threadIdx.x;
    if (e < EE) atomAddF(&deg[dst[e]], 1.0f);
}

__global__ void k_dinv(float* __restrict__ d) {
    int i = blockIdx.x * blockDim.x + threadIdx.x;
    if (i < NN) d[i] = rsqrtf(d[i]);   // deg >= 1 always
}

// h = x @ W_in^T + b_in   (K = 3, pure elementwise)
__global__ void k_inproj(const float* __restrict__ x, const float* __restrict__ Win,
                         const float* __restrict__ bin, float* __restrict__ h) {
    int i = blockIdx.x * blockDim.x + threadIdx.x;
    if (i >= NN * HH) return;
    int n = i >> 7, f = i & 127;
    const float* xr = x + n * 3;
    const float* wr = Win + f * 3;
    h[i] = fmaf(xr[0], wr[0], fmaf(xr[1], wr[1], fmaf(xr[2], wr[2], bin[f])));
}

// ---------------- WMMA GEMM: m[N,H] = h[N,H] @ W[H,H]^T ----------------
// block = 256 threads = 8 waves; wave w owns 16x16 output tile (rows blockIdx*16, cols w*16)
__global__ void __launch_bounds__(256) k_gemm_wmma(const float* __restrict__ A,
                                                   const float* __restrict__ W,
                                                   float* __restrict__ Mout) {
    const int lane = threadIdx.x & 31;
    const int wv   = threadIdx.x >> 5;    // 0..7 -> column tile
    const int half = lane >> 4;           // 0/1
    const int l16  = lane & 15;
    const int rowBase = blockIdx.x << 4;  // N = 6250*16 exactly, no remainder
    const int colBase = wv << 4;

    const float* arow = A + (size_t)(rowBase + l16) * HH;  // A row for this lane
    const float* wrow = W + (size_t)(colBase + l16) * HH;  // B col n -> W row n (B[k][n]=W[n][k])

    v8f acc = {};
#pragma unroll
    for (int kt = 0; kt < HH; kt += 32) {
        v16h af, bf;
#pragma unroll
        for (int v = 0; v < 8; ++v) {
            // A 16x32 f16 layout: lanes0-15 K=0..7 (V0-3), K=16..23 (V4-7); lanes16-31 +8
            int ka = kt + ((v & 3) << 1) + (half << 3) + ((v >> 2) << 4);
            // B 32x16 f16 layout: same K order as A (lanes0-15 K=0..7, 16..23; lanes16-31 +8)
            int kb = kt + ((v & 3) << 1) + (half << 3) + ((v >> 2) << 4);
            float2 ta = *(const float2*)(arow + ka);
            float2 tb = *(const float2*)(wrow + kb);
            af[2 * v]     = (_Float16)ta.x;
            af[2 * v + 1] = (_Float16)ta.y;
            bf[2 * v]     = (_Float16)tb.x;
            bf[2 * v + 1] = (_Float16)tb.y;
        }
        acc = __builtin_amdgcn_wmma_f32_16x16x32_f16(
            /*neg_a=*/false, af, /*neg_b=*/false, bf,
            /*c_mod=*/(short)0, acc, /*reuse_a=*/false, /*reuse_b=*/false);
    }
#pragma unroll
    for (int j = 0; j < 8; ++j) {
        int r = rowBase + (half << 3) + j;            // C/D: VGPR j = row j (+8 for upper lanes)
        Mout[(size_t)r * HH + colBase + l16] = acc[j];
    }
}

// agg init: self-loop contribution + bias:  agg[n][f] = m[n][f]*dinv[n]^2 + b_gcn[f]
__global__ void k_agg_init(const float* __restrict__ m, const float* __restrict__ dinv,
                           const float* __restrict__ bg, float* __restrict__ agg) {
    int i = blockIdx.x * blockDim.x + threadIdx.x;
    if (i >= NN * HH) return;
    int n = i >> 7, f = i & 127;
    float di = dinv[n];
    agg[i] = fmaf(m[i], di * di, bg[f]);
}

// edge scatter: 32-lane group per edge, 4 floats per lane (float4 gather, f32 atomics into L2)
__global__ void __launch_bounds__(256) k_scatter(const float* __restrict__ m,
                                                 const int* __restrict__ src,
                                                 const int* __restrict__ dst,
                                                 const float* __restrict__ dinv,
                                                 float* __restrict__ agg) {
    int grp  = threadIdx.x >> 5;
    int lane = threadIdx.x & 31;
    int e = blockIdx.x * 8 + grp;
    if (e >= EE) return;
    int s = src[e], d = dst[e];
    float w = dinv[s] * dinv[d];
    float4 v = ((const float4*)(m + (size_t)s * HH))[lane];
    float* ad = agg + (size_t)d * HH + (lane << 2);
    atomAddF(ad + 0, v.x * w);
    atomAddF(ad + 1, v.y * w);
    atomAddF(ad + 2, v.z * w);
    atomAddF(ad + 3, v.w * w);
}

// BN column stats: thread f accumulates sum/sumsq over a stride of rows
__global__ void __launch_bounds__(128) k_bnstats(const float* __restrict__ agg,
                                                 float* __restrict__ bnsum,
                                                 float* __restrict__ bnsq) {
    int f = threadIdx.x;
    float s = 0.0f, q = 0.0f;
    for (int n = blockIdx.x; n < NN; n += gridDim.x) {
        float v = agg[(size_t)n * HH + f];
        s += v;
        q = fmaf(v, v, q);
    }
    atomAddF(&bnsum[f], s);
    atomAddF(&bnsq[f], q);
}

// BN apply + residual + ReLU, h updated in place
__global__ void k_bnapply(const float* __restrict__ agg, float* __restrict__ h,
                          const float* __restrict__ bnsum, const float* __restrict__ bnsq,
                          const float* __restrict__ gamma, const float* __restrict__ beta) {
    int i = blockIdx.x * blockDim.x + threadIdx.x;
    if (i >= NN * HH) return;
    int f = i & 127;
    const float invN = 1.0f / (float)NN;
    float mu  = bnsum[f] * invN;
    float var = fmaf(-mu, mu, bnsq[f] * invN);
    float sc  = gamma[f] * rsqrtf(var + EPS_BN);
    float v   = fmaf(agg[i] - mu, sc, beta[f]) + h[i];
    h[i] = fmaxf(v, 0.0f);
}

// global mean+max pooling; h >= 0 post-ReLU so int-bits atomicMax is order-correct
__global__ void __launch_bounds__(128) k_pool(const float* __restrict__ h,
                                              const int* __restrict__ batch,
                                              float* __restrict__ psum,
                                              int* __restrict__ pmax,
                                              float* __restrict__ cnt) {
    int n = blockIdx.x;
    int f = threadIdx.x;
    int g = batch[n];
    float v = h[(size_t)n * HH + f];
    atomAddF(&psum[g * HH + f], v);
    atomicMax(&pmax[g * HH + f], __float_as_int(v));
    if (f == 0) atomAddF(&cnt[g], 1.0f);
}

// classifier MLP: one block per graph
__global__ void __launch_bounds__(128) k_classifier(const float* __restrict__ psum,
                                                    const int* __restrict__ pmax,
                                                    const float* __restrict__ cnt,
                                                    const float* __restrict__ Wc1, const float* __restrict__ bc1,
                                                    const float* __restrict__ Wc2, const float* __restrict__ bc2,
                                                    const float* __restrict__ Wc3, const float* __restrict__ bc3,
                                                    float* __restrict__ out) {
    __shared__ float p[2 * HH];
    __shared__ float o1[HH];
    __shared__ float o2[HH / 2];
    int g = blockIdx.x, f = threadIdx.x;
    float c = fmaxf(cnt[g], 1.0f);
    p[f]      = psum[g * HH + f] / c;
    p[HH + f] = __int_as_float(pmax[g * HH + f]);
    __syncthreads();
    {
        float a = bc1[f];
        const float* w = Wc1 + f * (2 * HH);
#pragma unroll 8
        for (int k = 0; k < 2 * HH; ++k) a = fmaf(w[k], p[k], a);
        o1[f] = fmaxf(a, 0.0f);
    }
    __syncthreads();
    if (f < HH / 2) {
        float a = bc2[f];
        const float* w = Wc2 + f * HH;
#pragma unroll 8
        for (int k = 0; k < HH; ++k) a = fmaf(w[k], o1[k], a);
        o2[f] = fmaxf(a, 0.0f);
    }
    __syncthreads();
    if (f < CC) {
        float a = bc3[f];
        const float* w = Wc3 + f * (HH / 2);
#pragma unroll 8
        for (int k = 0; k < HH / 2; ++k) a = fmaf(w[k], o2[k], a);
        out[g * CC + f] = a;
    }
}

extern "C" void kernel_launch(void* const* d_in, const int* in_sizes, int n_in,
                              void* d_out, int out_size, void* d_ws, size_t ws_size,
                              hipStream_t stream) {
    const float* x     = (const float*)d_in[0];
    const int*   ei    = (const int*)d_in[1];
    const int*   batch = (const int*)d_in[2];
    const float* Win   = (const float*)d_in[3];
    const float* bin   = (const float*)d_in[4];
    const float* Wg    = (const float*)d_in[5];
    const float* bg    = (const float*)d_in[6];
    const float* gamma = (const float*)d_in[7];
    const float* beta  = (const float*)d_in[8];
    const float* Wc1   = (const float*)d_in[9];
    const float* bc1   = (const float*)d_in[10];
    const float* Wc2   = (const float*)d_in[11];
    const float* bc2   = (const float*)d_in[12];
    const float* Wc3   = (const float*)d_in[13];
    const float* bc3   = (const float*)d_in[14];
    float* out = (float*)d_out;

    float* ws = (float*)d_ws;
    size_t o = 0;
    float* h    = ws + o; o += (size_t)NN * HH;
    float* m    = ws + o; o += (size_t)NN * HH;
    float* agg  = ws + o; o += (size_t)NN * HH;
    float* dinv = ws + o; o += NN;
    float* bnsum= ws + o; o += HH;        // bnsum, bnsq contiguous (zeroed together)
    float* bnsq = ws + o; o += HH;
    float* psum = ws + o; o += (size_t)GG * HH;   // psum, pmax, cnt contiguous
    int*   pmax = (int*)(ws + o); o += (size_t)GG * HH;
    float* cnt  = ws + o; o += GG;

    const int* src = ei;        // edge_index[0]
    const int* dst = ei + EE;   // edge_index[1]

    // degree (with self loop) -> dinv = rsqrt(deg)
    k_init_deg<<<(NN + 255) / 256, 256, 0, stream>>>(dinv);
    k_count_deg<<<(EE + 255) / 256, 256, 0, stream>>>(dst, dinv);
    k_dinv<<<(NN + 255) / 256, 256, 0, stream>>>(dinv);

    // input projection
    k_inproj<<<(NN * HH + 255) / 256, 256, 0, stream>>>(x, Win, bin, h);

    for (int l = 0; l < LL; ++l) {
        k_gemm_wmma<<<NN / 16, 256, 0, stream>>>(h, Wg + (size_t)l * HH * HH, m);
        k_agg_init<<<(NN * HH + 255) / 256, 256, 0, stream>>>(m, dinv, bg + l * HH, agg);
        k_scatter<<<EE / 8, 256, 0, stream>>>(m, src, dst, dinv, agg);
        k_zero<<<1, 256, 0, stream>>>(bnsum, 2 * HH);
        k_bnstats<<<240, 128, 0, stream>>>(agg, bnsum, bnsq);
        k_bnapply<<<(NN * HH + 255) / 256, 256, 0, stream>>>(agg, h, bnsum, bnsq,
                                                             gamma + l * HH, beta + l * HH);
    }

    // pooling + classifier
    k_zero<<<(2 * GG * HH + GG + 255) / 256, 256, 0, stream>>>(psum, 2 * GG * HH + GG);
    k_pool<<<NN, 128, 0, stream>>>(h, batch, psum, pmax, cnt);
    k_classifier<<<GG, 128, 0, stream>>>(psum, pmax, cnt, Wc1, bc1, Wc2, bc2, Wc3, bc3, out);
}
